// DecoderLayer_44676249813143
// MI455X (gfx1250) — matrix operand order repaired and checked by reading, not compiled, not run on hardware
//
#include <hip/hip_runtime.h>

// ---------------------------------------------------------------------------
// Decoder layer for MI455X (gfx1250, wave32, WMMA).
// All matmuls use v_wmma_f32_16x16x32_f16 (f16 in, f32 accumulate).
// GEMM: 64x32 output tile per wave (4x2 accumulators = 64 VGPRs, 6 fragments
// = 48 VGPRs) -> 8 WMMAs per 6KB of fragment loads with no spilling.
// ---------------------------------------------------------------------------

typedef __attribute__((ext_vector_type(16))) _Float16 v16h;
typedef __attribute__((ext_vector_type(8)))  float    v8f;

#define LN_EPS 1e-5f

static __device__ __forceinline__ v8f zero8() {
    v8f z = {0.f, 0.f, 0.f, 0.f, 0.f, 0.f, 0.f, 0.f};
    return z;
}

// Load a 16x32 A-fragment (row-major source, leading dim ld) following the
// documented CDNA5 16-bit A-matrix layout:
//   lane m = lane&15, hb8 = (lane>=16)?8:0
//   elements 0..7  -> k = hb8 + 0..7      (16 contiguous bytes)
//   elements 8..15 -> k = hb8 + 16..23    (16 contiguous bytes)
static __device__ __forceinline__ v16h load_frag_a(const _Float16* __restrict__ base,
                                                   int ld, int lane) {
    const int m   = lane & 15;
    const int hb8 = (lane >> 4) << 3;
    const _Float16* p = base + (size_t)m * ld + hb8;
    v16h r;
    ((uint4*)&r)[0] = *(const uint4*)(p);
    ((uint4*)&r)[1] = *(const uint4*)(p + 16);
    return r;
}

// Load a 32x16 B-fragment where B[k][n] = S[n][k] (S row-major, leading dim ld):
//   lane n = lane&15, koff = (lane>=16)?8:0, elements 0..7 -> k = koff + 0..7,
//   elements 8..15 -> k = koff + 16..23 (same k order as the A-fragment).
static __device__ __forceinline__ v16h load_frag_b(const _Float16* __restrict__ Srow0,
                                                   int ld, int lane) {
    const int n    = lane & 15;
    const int koff = (lane >> 4) << 3;
    const _Float16* p = Srow0 + (size_t)n * ld + koff;
    v16h r;
    ((uint4*)&r)[0] = *(const uint4*)(p);
    ((uint4*)&r)[1] = *(const uint4*)(p + 16);
    return r;
}

static __device__ __forceinline__ v8f wmma_f16(v16h a, v16h b, v8f c) {
    return __builtin_amdgcn_wmma_f32_16x16x32_f16(false, a, false, b, (short)0, c,
                                                  false, false);
}

// ---------------------------------------------------------------------------
// fp32 -> f16 conversion (optionally scaled)
// ---------------------------------------------------------------------------
__global__ void cvt_f32_f16_kernel(const float* __restrict__ src,
                                   _Float16* __restrict__ dst,
                                   size_t n, float scale) {
    size_t i = (size_t)blockIdx.x * blockDim.x + threadIdx.x;
    if (i < n) dst[i] = (_Float16)(src[i] * scale);
}

// ---------------------------------------------------------------------------
// GEMM: C[m,n] = sum_k A[m,k] * W[n,k]  (+ bias[n]) (+ SELU)
// A: MxK f16 row-major, W: NxK f16 row-major (torch Linear weight).
// Each wave computes a 64x32 tile: 4x2 WMMA accumulators, 4x2 operand reuse.
// outF (f32) and/or outH (f16, scaled by hscale) may be null.
// ---------------------------------------------------------------------------
__global__ void __launch_bounds__(256, 1)
gemm_f16_wmma_kernel(const _Float16* __restrict__ A,
                     const _Float16* __restrict__ W,
                     const float* __restrict__ bias,
                     float* __restrict__ outF,
                     _Float16* __restrict__ outH,
                     int M, int N, int K,
                     float hscale, int act) {
    const int lane = threadIdx.x & 31;
    const int wave = blockIdx.x * (blockDim.x >> 5) + (threadIdx.x >> 5);
    const int tilesN = N >> 5;   // 32-wide N tiles
    const int tilesM = M >> 6;   // 64-tall M tiles
    if (wave >= tilesM * tilesN) return;   // wave-uniform: EXEC stays all-ones
    const int m0 = (wave / tilesN) << 6;
    const int n0 = (wave % tilesN) << 5;

    v8f acc[4][2];
#pragma unroll
    for (int i = 0; i < 4; ++i)
#pragma unroll
        for (int j = 0; j < 2; ++j) acc[i][j] = zero8();

    for (int k0 = 0; k0 < K; k0 += 32) {
        v16h a[4], b[2];
#pragma unroll
        for (int i = 0; i < 4; ++i)
            a[i] = load_frag_a(A + (size_t)(m0 + i * 16) * K + k0, K, lane);
#pragma unroll
        for (int j = 0; j < 2; ++j)
            b[j] = load_frag_b(W + (size_t)(n0 + j * 16) * K + k0, K, lane);
#pragma unroll
        for (int i = 0; i < 4; ++i)
#pragma unroll
            for (int j = 0; j < 2; ++j)
                acc[i][j] = wmma_f16(a[i], b[j], acc[i][j]);
    }

    const int col16 = lane & 15;
    const int hb8   = (lane >> 4) << 3;
#pragma unroll
    for (int i = 0; i < 4; ++i) {
#pragma unroll
        for (int j = 0; j < 2; ++j) {
            const int nn = n0 + j * 16 + col16;
            float bv = bias ? bias[nn] : 0.f;
#pragma unroll
            for (int r = 0; r < 8; ++r) {
                const int mm = m0 + i * 16 + r + hb8;
                float v = acc[i][j][r] + bv;
                if (act == 1) {  // SELU
                    v = (v > 0.f) ? 1.0507009873554805f * v
                                  : 1.0507009873554805f * 1.6732632423543772f *
                                        (__expf(v) - 1.f);
                }
                const size_t o = (size_t)mm * N + nn;
                if (outF) outF[o] = v;
                if (outH) outH[o] = (_Float16)(v * hscale);
            }
        }
    }
}

// ---------------------------------------------------------------------------
// Flash attention (causal). One wave per (b, h, 16-query block).
// Q pre-scaled by 1/sqrt(dh). Q,K: [B,L,H,dh] f16. Vt: [B,H,dh,L] f16.
// Ctx out: [B,L,H*dh] f32. dh = 64, key blocks of 32.
// ---------------------------------------------------------------------------
__global__ void attn_flash_kernel(const _Float16* __restrict__ Q,
                                  const _Float16* __restrict__ Kx,
                                  const _Float16* __restrict__ Vt,
                                  float* __restrict__ Ctx,
                                  int Bz, int H, int L, int dh) {
    __shared__ _Float16 lds[8 * 16 * 32];  // 1 KB per wave (16x32 f16 P tile)
    const int lane = threadIdx.x & 31;
    const int wv   = threadIdx.x >> 5;
    const int wave = blockIdx.x * 8 + wv;
    const int QB   = L >> 4;
    if (wave >= Bz * H * QB) return;       // wave-uniform
    const int qb = wave % QB;
    const int h  = (wave / QB) % H;
    const int b  = wave / (QB * H);
    const int q0 = qb << 4;
    const int D  = H * dh;

    const _Float16* Qb = Q  + (size_t)b * L * D + (size_t)h * dh;
    const _Float16* Kb = Kx + (size_t)b * L * D + (size_t)h * dh;
    const _Float16* Vb = Vt + ((size_t)(b * H + h) * dh) * L;

    const int mrow = lane & 15;
    const int hb   = lane >> 4;
    const int hb8  = hb << 3;
    _Float16* lp = &lds[wv * 512];

    // Q fragments over dh=64 (two K-chunks of 32)
    v16h qa[2];
#pragma unroll
    for (int c = 0; c < 2; ++c)
        qa[c] = load_frag_a(Qb + (size_t)q0 * D + c * 32, D, lane);

    v8f ctx[4] = {zero8(), zero8(), zero8(), zero8()};
    float mrun[8], lrun[8];
#pragma unroll
    for (int r = 0; r < 8; ++r) { mrun[r] = -1e30f; lrun[r] = 0.f; }

    const int nb = (q0 + 15) / 32 + 1;
    for (int kb = 0; kb < nb; ++kb) {
        const int key0 = kb << 5;

        // scores: S = Q * K^T  (16 x 32 keys, two 16-col C tiles)
        v8f s[2] = {zero8(), zero8()};
#pragma unroll
        for (int c = 0; c < 2; ++c)
#pragma unroll
            for (int kc = 0; kc < 2; ++kc)
                s[c] = wmma_f16(qa[kc],
                                load_frag_b(Kb + (size_t)(key0 + c * 16) * D + kc * 32,
                                            D, lane),
                                s[c]);

        // causal mask on partial blocks
        if (key0 + 31 > q0) {
#pragma unroll
            for (int c = 0; c < 2; ++c) {
                const int key = key0 + c * 16 + mrow;
#pragma unroll
                for (int r = 0; r < 8; ++r)
                    if (key > q0 + r + hb8) s[c][r] = -1e30f;
            }
        }

        // online softmax; row r lives in lanes of the same 16-lane group
#pragma unroll
        for (int r = 0; r < 8; ++r) {
            float mx = fmaxf(s[0][r], s[1][r]);
#pragma unroll
            for (int off = 8; off >= 1; off >>= 1)
                mx = fmaxf(mx, __shfl_xor(mx, off, 16));
            const float mnew = fmaxf(mrun[r], mx);
            const float p0 = __expf(s[0][r] - mnew);
            const float p1 = __expf(s[1][r] - mnew);
            float rs = p0 + p1;
#pragma unroll
            for (int off = 8; off >= 1; off >>= 1)
                rs += __shfl_xor(rs, off, 16);
            const float corr = __expf(mrun[r] - mnew);
            lrun[r] = lrun[r] * corr + rs;
            mrun[r] = mnew;
#pragma unroll
            for (int j = 0; j < 4; ++j) ctx[j][r] *= corr;
            // stash P (C-layout) into LDS
            lp[(r + hb8) * 32 + mrow]      = (_Float16)p0;
            lp[(r + hb8) * 32 + 16 + mrow] = (_Float16)p1;
        }

        // read P back in A-fragment layout (LDS ops in-order within wave)
        v16h pa;
        const _Float16* pr = lp + mrow * 32 + hb8;
        ((uint4*)&pa)[0] = *(const uint4*)(pr);
        ((uint4*)&pa)[1] = *(const uint4*)(pr + 16);

        // ctx += P * V ; Vt rows are head-dims so B-fragments are contiguous
#pragma unroll
        for (int j = 0; j < 4; ++j)
            ctx[j] = wmma_f16(pa,
                              load_frag_b(Vb + (size_t)(j * 16) * L + key0, L, lane),
                              ctx[j]);
    }

    float* Ob = Ctx + (size_t)b * L * D + (size_t)h * dh;
#pragma unroll
    for (int j = 0; j < 4; ++j)
#pragma unroll
        for (int r = 0; r < 8; ++r)
            Ob[(size_t)(q0 + r + hb8) * D + j * 16 + mrow] = ctx[j][r] / lrun[r];
}

// ---------------------------------------------------------------------------
// Vt[b][h][d][l] = V[b][l][h][d]
// ---------------------------------------------------------------------------
__global__ void transpose_v_kernel(const _Float16* __restrict__ V,
                                   _Float16* __restrict__ Vt,
                                   int Bz, int H, int L, int dh) {
    size_t idx = (size_t)blockIdx.x * blockDim.x + threadIdx.x;
    const size_t total = (size_t)Bz * H * L * dh;
    if (idx >= total) return;
    int l = (int)(idx % L); size_t t = idx / L;
    int d = (int)(t % dh);  t /= dh;
    int h = (int)(t % H);   int b = (int)(t / H);
    Vt[idx] = V[(((size_t)b * L + l) * H + h) * dh + d];
}

// ---------------------------------------------------------------------------
// y = LN(a + c) * g + beta ; one 256-thread block per row, D = 1024.
// ---------------------------------------------------------------------------
__global__ void resid_ln_kernel(const float* __restrict__ a,
                                const float* __restrict__ c,
                                const float* __restrict__ g,
                                const float* __restrict__ beta,
                                float* __restrict__ outF,
                                _Float16* __restrict__ outH, int D) {
    const int row = blockIdx.x;
    const float* ar = a + (size_t)row * D;
    const float* cr = c + (size_t)row * D;
    float vals[4], s = 0.f, s2 = 0.f;
#pragma unroll
    for (int i = 0; i < 4; ++i) {
        const int idx = threadIdx.x + i * 256;
        const float v = ar[idx] + cr[idx];
        vals[i] = v; s += v; s2 += v * v;
    }
#pragma unroll
    for (int off = 16; off >= 1; off >>= 1) {
        s  += __shfl_xor(s,  off, 32);
        s2 += __shfl_xor(s2, off, 32);
    }
    __shared__ float ws[8], ws2[8];
    if ((threadIdx.x & 31) == 0) {
        ws[threadIdx.x >> 5]  = s;
        ws2[threadIdx.x >> 5] = s2;
    }
    __syncthreads();
    s = 0.f; s2 = 0.f;
#pragma unroll
    for (int k = 0; k < 8; ++k) { s += ws[k]; s2 += ws2[k]; }
    const float mean = s / D;
    const float var  = s2 / D - mean * mean;
    const float rinv = rsqrtf(var + LN_EPS);
#pragma unroll
    for (int i = 0; i < 4; ++i) {
        const int idx = threadIdx.x + i * 256;
        const float y = (vals[i] - mean) * rinv * g[idx] + beta[idx];
        const size_t o = (size_t)row * D + idx;
        if (outF) outF[o] = y;
        if (outH) outH[o] = (_Float16)y;
    }
}

// ---------------------------------------------------------------------------
// Host orchestration
// ---------------------------------------------------------------------------
extern "C" void kernel_launch(void* const* d_in, const int* in_sizes, int n_in,
                              void* d_out, int out_size, void* d_ws, size_t ws_size,
                              hipStream_t stream) {
    (void)in_sizes; (void)n_in; (void)out_size; (void)ws_size;
    const int Bz = 4, L = 1024, D = 1024, H = 16, dh = 64, Dff = 4096;
    const size_t M = (size_t)Bz * L;

    const float* X    = (const float*)d_in[0];
    const float* wq   = (const float*)d_in[1];
    const float* wk   = (const float*)d_in[2];
    const float* wv   = (const float*)d_in[3];
    const float* ln1g = (const float*)d_in[4];
    const float* ln1b = (const float*)d_in[5];
    const float* w1   = (const float*)d_in[6];
    const float* b1   = (const float*)d_in[7];
    const float* w2   = (const float*)d_in[8];
    const float* b2   = (const float*)d_in[9];
    const float* ln2g = (const float*)d_in[10];
    const float* ln2b = (const float*)d_in[11];
    float* out = (float*)d_out;

    char* ws = (char*)d_ws;
    size_t off = 0;
    auto alloc = [&](size_t bytes) -> void* {
        void* p = ws + off;
        off = (off + bytes + 255) & ~(size_t)255;
        return p;
    };
    _Float16* Xh  = (_Float16*)alloc(M * D * 2);
    _Float16* Wqh = (_Float16*)alloc((size_t)D * D * 2);
    _Float16* Wkh = (_Float16*)alloc((size_t)D * D * 2);
    _Float16* Wvh = (_Float16*)alloc((size_t)D * D * 2);
    _Float16* W1h = (_Float16*)alloc((size_t)Dff * D * 2);
    _Float16* W2h = (_Float16*)alloc((size_t)D * Dff * 2);
    _Float16* Qh  = (_Float16*)alloc(M * D * 2);
    _Float16* Kh  = (_Float16*)alloc(M * D * 2);
    _Float16* Vh  = (_Float16*)alloc(M * D * 2);
    _Float16* Vth = (_Float16*)alloc(M * D * 2);
    float*    Ctx = (float*)alloc(M * D * 4);
    float*    X1  = (float*)alloc(M * D * 4);
    _Float16* X1h = (_Float16*)alloc(M * D * 2);
    _Float16* Hh  = (_Float16*)alloc(M * Dff * 2);
    float*    Y2  = (float*)alloc(M * D * 4);

    const int T = 256;
    auto cgrid = [](size_t n) { return (unsigned)((n + 255) / 256); };

    // fp32 -> f16 conversions
    cvt_f32_f16_kernel<<<cgrid(M * D), T, 0, stream>>>(X, Xh, M * D, 1.f);
    cvt_f32_f16_kernel<<<cgrid((size_t)D * D), T, 0, stream>>>(wq, Wqh, (size_t)D * D, 1.f);
    cvt_f32_f16_kernel<<<cgrid((size_t)D * D), T, 0, stream>>>(wk, Wkh, (size_t)D * D, 1.f);
    cvt_f32_f16_kernel<<<cgrid((size_t)D * D), T, 0, stream>>>(wv, Wvh, (size_t)D * D, 1.f);
    cvt_f32_f16_kernel<<<cgrid((size_t)Dff * D), T, 0, stream>>>(w1, W1h, (size_t)Dff * D, 1.f);
    cvt_f32_f16_kernel<<<cgrid((size_t)D * Dff), T, 0, stream>>>(w2, W2h, (size_t)D * Dff, 1.f);

    auto ggrid = [](size_t m, size_t n) {
        return (unsigned)(((m / 64) * (n / 32) + 7) / 8);
    };
    // QKV projections (Q pre-scaled by 1/sqrt(dh) = 0.125)
    gemm_f16_wmma_kernel<<<ggrid(M, D), T, 0, stream>>>(
        Xh, Wqh, nullptr, nullptr, Qh, (int)M, D, D, 0.125f, 0);
    gemm_f16_wmma_kernel<<<ggrid(M, D), T, 0, stream>>>(
        Xh, Wkh, nullptr, nullptr, Kh, (int)M, D, D, 1.f, 0);
    gemm_f16_wmma_kernel<<<ggrid(M, D), T, 0, stream>>>(
        Xh, Wvh, nullptr, nullptr, Vh, (int)M, D, D, 1.f, 0);

    transpose_v_kernel<<<cgrid(M * D), T, 0, stream>>>(Vh, Vth, Bz, H, L, dh);

    // causal flash attention -> Ctx (f32)
    attn_flash_kernel<<<(unsigned)((Bz * H * (L / 16) + 7) / 8), T, 0, stream>>>(
        Qh, Kh, Vth, Ctx, Bz, H, L, dh);

    // X1 = LN(X + Ctx)
    resid_ln_kernel<<<(unsigned)M, T, 0, stream>>>(X, Ctx, ln1g, ln1b, X1, X1h, D);

    // H = selu(X1 @ w1^T + b1)
    gemm_f16_wmma_kernel<<<ggrid(M, Dff), T, 0, stream>>>(
        X1h, W1h, b1, nullptr, Hh, (int)M, Dff, D, 1.f, 1);
    // Y2 = H @ w2^T + b2
    gemm_f16_wmma_kernel<<<ggrid(M, D), T, 0, stream>>>(
        Hh, W2h, b2, Y2, nullptr, (int)M, D, Dff, 1.f, 0);

    // out = LN(X1 + Y2)
    resid_ln_kernel<<<(unsigned)M, T, 0, stream>>>(X1, Y2, ln2g, ln2b, out, nullptr, D);
}
